// PeriodicPointNet_609885356793
// MI455X (gfx1250) — compile-verified, matrix operand order checked by reading, not run on hardware
//
#include <hip/hip_runtime.h>
#include <hip/hip_bf16.h>

typedef __attribute__((ext_vector_type(16))) _Float16 v16h;
typedef __attribute__((ext_vector_type(8)))  float    v8f;

// --------------------------------------------------------------------------
// WMMA 16x16x32 f16 fragment index map (CDNA5 ISA 7.12.2, 16-bit A-matrix):
// lane L (0..15): rows M=L, K = {2p,2p+1} for vgpr p<4, {16+2(p-4),..} p>=4
// lane L+16: same rows, K offset +8.  B fragment mirrors with lane = N col.
// --------------------------------------------------------------------------
__device__ __forceinline__ int frag_k(int p, int lo, int hi) {
    return (p < 4) ? (2 * p + lo + 8 * hi) : (16 + 2 * (p - 4) + lo + 8 * hi);
}

// Load A fragment (16 rows x 32 K) from LDS, row-major halves, ldm = row stride.
__device__ __forceinline__ v16h load_afrag_lds(const _Float16* lds, int ldm,
                                               int mbase, int kbase) {
    int lane = threadIdx.x & 31;
    int hi = lane >> 4;
    const _Float16* row = lds + (mbase + (lane & 15)) * ldm + kbase;
    v16h r;
#pragma unroll
    for (int p = 0; p < 8; ++p) {
        int k = frag_k(p, 0, hi);
        r[2 * p]     = row[k];
        r[2 * p + 1] = row[k + 1];
    }
    return r;
}

// Load pre-swizzled B fragment: layout [kt][nt][lane][16] halves, 32B/lane.
__device__ __forceinline__ v16h load_wfrag(const _Float16* F, int ntiles,
                                           int kt, int nt) {
    int lane = threadIdx.x & 31;
    const _Float16* p = F + ((((size_t)kt * ntiles + nt) * 32 + lane) << 4);
    return *(const v16h*)p;
}

// Store C tile to LDS as f16 with bias + ReLU.
__device__ __forceinline__ void store_ctile_lds(_Float16* lds, int ldm, int mbase,
                                                int nbase, v8f acc, const float* bias) {
    int lane = threadIdx.x & 31;
    int hi = lane >> 4;
    int n = nbase + (lane & 15);
    float bv = bias[n];
#pragma unroll
    for (int r = 0; r < 8; ++r) {
        float v = acc[r] + bv;
        v = v > 0.f ? v : 0.f;
        lds[(mbase + r + 8 * hi) * ldm + n] = (_Float16)v;
    }
}

// --------------------------------------------------------------------------
// Weight converter: f32 [K][N] row-major  ->  f16 B-fragment layout with
// K zero-padded to Kpad (multiple of 32).
// --------------------------------------------------------------------------
__global__ void prep_weight(const float* __restrict__ W, _Float16* __restrict__ F,
                            int K, int N, int Kpad) {
    int ntiles = N >> 4;
    int total = Kpad * N;
    for (int idx = blockIdx.x * blockDim.x + threadIdx.x; idx < total;
         idx += gridDim.x * blockDim.x) {
        int h = idx & 15;
        int lane = (idx >> 4) & 31;
        int t = idx >> 9;                   // kt*ntiles + nt
        int nt = t % ntiles, kt = t / ntiles;
        int p = h >> 1, lo = h & 1, hi = lane >> 4;
        int k = kt * 32 + frag_k(p, lo, hi);
        int n = nt * 16 + (lane & 15);
        float v = (k < K) ? W[(size_t)k * N + n] : 0.f;
        F[idx] = (_Float16)v;
    }
}

__global__ void zero_out(float* out, int n) {
    int i = blockIdx.x * blockDim.x + threadIdx.x;
    if (i < n) out[i] = 0.f;
}

// --------------------------------------------------------------------------
// Per-query periodic distances + top-64 selection (shared by both SA layers).
// One 256-thread block per (b,q).
// --------------------------------------------------------------------------
__global__ void topk_kernel(const float* __restrict__ frac,
                            const float* __restrict__ trans,
                            int* __restrict__ nbr, float* __restrict__ ndist) {
    int b = blockIdx.x >> 9, q = blockIdx.x & 511;
    __shared__ float D[512];
    __shared__ float rv[256];
    __shared__ int   ri[256];
    __shared__ float T[9], fq[3];
    int tid = threadIdx.x;
    if (tid < 9) T[tid] = trans[b * 9 + tid];
    if (tid < 3) fq[tid] = frac[((size_t)b * 512 + q) * 3 + tid];
    __syncthreads();
    for (int p = tid; p < 512; p += 256) {
        const float* fp = frac + ((size_t)b * 512 + p) * 3;
        float d0 = fp[0] - fq[0], d1 = fp[1] - fq[1], d2 = fp[2] - fq[2];
        float c0 = d0 * T[0] + d1 * T[3] + d2 * T[6];
        float c1 = d0 * T[1] + d1 * T[4] + d2 * T[7];
        float c2 = d0 * T[2] + d1 * T[5] + d2 * T[8];
        D[p] = c0 * c0 + c1 * c1 + c2 * c2;
    }
    __syncthreads();
    size_t base = ((size_t)b * 512 + q) * 64;
    for (int it = 0; it < 64; ++it) {
        float v = D[tid]; int i = tid;
        float v2 = D[tid + 256];
        if (v2 < v) { v = v2; i = tid + 256; }
        rv[tid] = v; ri[tid] = i;
        __syncthreads();
        for (int s = 128; s > 0; s >>= 1) {
            if (tid < s && rv[tid + s] < rv[tid]) {
                rv[tid] = rv[tid + s]; ri[tid] = ri[tid + s];
            }
            __syncthreads();
        }
        if (tid == 0) {
            nbr[base + it] = ri[0];
            ndist[base + it] = rv[0];
            D[ri[0]] = 3.0e38f;
        }
        __syncthreads();
    }
}

// --------------------------------------------------------------------------
// Fused PointConv SA layer: gather -> 3-layer MLP (WMMA f16) -> masked max.
// One 128-thread block (4 waves) per query; wave w owns edge rows 16w..16w+15.
// --------------------------------------------------------------------------
template <typename TIN, int CIN, int KPAD0, int H1, int H2, int COUT>
__global__ void __launch_bounds__(128)
sa_kernel(const TIN* __restrict__ feat, const float* __restrict__ pos,
          const int* __restrict__ nbr, const float* __restrict__ ndist,
          const float* __restrict__ scale, float radius,
          const _Float16* __restrict__ W0f, const float* __restrict__ b0,
          const _Float16* __restrict__ W1f, const float* __restrict__ b1,
          const _Float16* __restrict__ W2f, const float* __restrict__ b2,
          _Float16* __restrict__ out) {
    int b = blockIdx.x >> 9, q = blockIdx.x & 511;
    __shared__ _Float16 s_msg[64 * KPAD0];
    __shared__ _Float16 s_h1[64 * H1];
    __shared__ _Float16 s_h2[64 * H2];
    __shared__ float s_red[4 * COUT];
    __shared__ int s_nbr[64];
    __shared__ unsigned s_vm[2];
    int tid = threadIdx.x, w = tid >> 5;
    size_t nb = ((size_t)b * 512 + q) * 64;

    if (tid < 2) s_vm[tid] = 0u;
    __syncthreads();
    if (tid < 64) {
        int j = nbr[nb + tid];
        s_nbr[tid] = j;
        float re = radius / scale[b];
        if (ndist[nb + tid] < re * re) atomicOr(&s_vm[tid >> 5], 1u << (tid & 31));
    }
    __syncthreads();

    // Gather msg = [x_j , pos_j - pos_q , 0-pad] as f16.
    const float* pq = pos + ((size_t)b * 512 + q) * 3;
    for (int idx = tid; idx < 64 * KPAD0; idx += 128) {
        int e = idx / KPAD0, c = idx - e * KPAD0;
        int j = s_nbr[e];
        float v = 0.f;
        if (c < CIN)            v = (float)feat[((size_t)b * 512 + j) * CIN + c];
        else if (c < CIN + 3)   v = pos[((size_t)b * 512 + j) * 3 + (c - CIN)] - pq[c - CIN];
        s_msg[idx] = (_Float16)v;
    }
    __syncthreads();

    // Layer 0: KPAD0 -> H1
    for (int nt = 0; nt < H1 / 16; ++nt) {
        v8f acc = {};
#pragma unroll
        for (int kt = 0; kt < KPAD0 / 32; ++kt) {
            v16h a = load_afrag_lds(s_msg, KPAD0, 16 * w, kt * 32);
            v16h bb = load_wfrag(W0f, H1 / 16, kt, nt);
            acc = __builtin_amdgcn_wmma_f32_16x16x32_f16(false, a, false, bb,
                                                         (short)0, acc, false, false);
        }
        store_ctile_lds(s_h1, H1, 16 * w, nt * 16, acc, b0);
    }
    __syncthreads();

    // Layer 1: H1 -> H2
    for (int nt = 0; nt < H2 / 16; ++nt) {
        v8f acc = {};
#pragma unroll
        for (int kt = 0; kt < H1 / 32; ++kt) {
            v16h a = load_afrag_lds(s_h1, H1, 16 * w, kt * 32);
            v16h bb = load_wfrag(W1f, H2 / 16, kt, nt);
            acc = __builtin_amdgcn_wmma_f32_16x16x32_f16(false, a, false, bb,
                                                         (short)0, acc, false, false);
        }
        store_ctile_lds(s_h2, H2, 16 * w, nt * 16, acc, b1);
    }
    __syncthreads();

    // Layer 2: H2 -> COUT, fused radius mask + row max.
    unsigned vm0 = s_vm[0], vm1 = s_vm[1];
    int lane = tid & 31, hi = lane >> 4;
    for (int nt = 0; nt < COUT / 16; ++nt) {
        v8f acc = {};
#pragma unroll
        for (int kt = 0; kt < H2 / 32; ++kt) {
            v16h a = load_afrag_lds(s_h2, H2, 16 * w, kt * 32);
            v16h bb = load_wfrag(W2f, COUT / 16, kt, nt);
            acc = __builtin_amdgcn_wmma_f32_16x16x32_f16(false, a, false, bb,
                                                         (short)0, acc, false, false);
        }
        int n = nt * 16 + (lane & 15);
        float bv = b2[n];
        float mx = -3.0e38f;
#pragma unroll
        for (int r = 0; r < 8; ++r) {
            float v = acc[r] + bv;
            v = v > 0.f ? v : 0.f;
            int e = 16 * w + r + 8 * hi;
            bool val = (e < 32) ? ((vm0 >> e) & 1u) : ((vm1 >> (e - 32)) & 1u);
            if (val) mx = fmaxf(mx, v);
        }
        mx = fmaxf(mx, __shfl_xor(mx, 16, 32));
        if (lane < 16) s_red[w * COUT + n] = mx;
    }
    __syncthreads();

    bool anyv = (vm0 | vm1) != 0u;
    for (int n = tid; n < COUT; n += 128) {
        float m = fmaxf(fmaxf(s_red[n], s_red[COUT + n]),
                        fmaxf(s_red[2 * COUT + n], s_red[3 * COUT + n]));
        out[((size_t)b * 512 + q) * COUT + n] = (_Float16)(anyv ? m : 0.f);
    }
}

// --------------------------------------------------------------------------
// mlp3 (259->256->512->1024) + global max pool. One block per 16-point tile,
// waves split N-tiles; final max combined across blocks via atomicMax on
// float bits (valid: post-ReLU values >= 0, d_out zeroed first).
// --------------------------------------------------------------------------
__global__ void __launch_bounds__(128)
mlp3_kernel(const _Float16* __restrict__ x2, const float* __restrict__ pos,
            const _Float16* __restrict__ W0f, const float* __restrict__ b0,
            const _Float16* __restrict__ W1f, const float* __restrict__ b1,
            const _Float16* __restrict__ W2f, const float* __restrict__ b2,
            float* __restrict__ out) {
    int b = blockIdx.x >> 5, q0 = (blockIdx.x & 31) * 16;
    __shared__ _Float16 s_in[16 * 288];
    __shared__ _Float16 s_h1[16 * 256];
    __shared__ _Float16 s_h2[16 * 512];
    int tid = threadIdx.x, w = tid >> 5;
    int lane = tid & 31;

    for (int idx = tid; idx < 16 * 288; idx += 128) {
        int m = idx / 288, c = idx - m * 288;
        size_t p = (size_t)b * 512 + q0 + m;
        float v = 0.f;
        if (c < 256)       v = (float)x2[p * 256 + c];
        else if (c < 259)  v = pos[p * 3 + (c - 256)];
        s_in[idx] = (_Float16)v;
    }
    __syncthreads();

    // L1: 288 -> 256 (16 ntiles split over 4 waves)
    for (int nt = w; nt < 16; nt += 4) {
        v8f acc = {};
#pragma unroll
        for (int kt = 0; kt < 9; ++kt) {
            v16h a = load_afrag_lds(s_in, 288, 0, kt * 32);
            v16h bb = load_wfrag(W0f, 16, kt, nt);
            acc = __builtin_amdgcn_wmma_f32_16x16x32_f16(false, a, false, bb,
                                                         (short)0, acc, false, false);
        }
        store_ctile_lds(s_h1, 256, 0, nt * 16, acc, b0);
    }
    __syncthreads();

    // L2: 256 -> 512
    for (int nt = w; nt < 32; nt += 4) {
        v8f acc = {};
#pragma unroll
        for (int kt = 0; kt < 8; ++kt) {
            v16h a = load_afrag_lds(s_h1, 256, 0, kt * 32);
            v16h bb = load_wfrag(W1f, 32, kt, nt);
            acc = __builtin_amdgcn_wmma_f32_16x16x32_f16(false, a, false, bb,
                                                         (short)0, acc, false, false);
        }
        store_ctile_lds(s_h2, 512, 0, nt * 16, acc, b1);
    }
    __syncthreads();

    // L3: 512 -> 1024, row max over 16 points, global atomic max.
    for (int nt = w; nt < 64; nt += 4) {
        v8f acc = {};
#pragma unroll
        for (int kt = 0; kt < 16; ++kt) {
            v16h a = load_afrag_lds(s_h2, 512, 0, kt * 32);
            v16h bb = load_wfrag(W2f, 64, kt, nt);
            acc = __builtin_amdgcn_wmma_f32_16x16x32_f16(false, a, false, bb,
                                                         (short)0, acc, false, false);
        }
        int n = nt * 16 + (lane & 15);
        float bv = b2[n];
        float mx = 0.f;
#pragma unroll
        for (int r = 0; r < 8; ++r) {
            float v = acc[r] + bv;
            mx = fmaxf(mx, v > 0.f ? v : 0.f);
        }
        mx = fmaxf(mx, __shfl_xor(mx, 16, 32));
        if (lane < 16) atomicMax((int*)&out[(size_t)b * 1024 + n], __float_as_int(mx));
    }
}

// --------------------------------------------------------------------------
extern "C" void kernel_launch(void* const* d_in, const int* in_sizes, int n_in,
                              void* d_out, int out_size, void* d_ws, size_t ws_size,
                              hipStream_t stream) {
    const float* x     = (const float*)d_in[0];   // [16,512,98]
    const float* pos   = (const float*)d_in[1];   // [16,512,3]
    const float* frac  = (const float*)d_in[2];   // [16,512,3]
    const float* trans = (const float*)d_in[3];   // [16,3,3]
    const float* scale = (const float*)d_in[4];   // [16]
    const float* W10 = (const float*)d_in[5],  *B10 = (const float*)d_in[6];
    const float* W11 = (const float*)d_in[7],  *B11 = (const float*)d_in[8];
    const float* W12 = (const float*)d_in[9],  *B12 = (const float*)d_in[10];
    const float* W20 = (const float*)d_in[11], *B20 = (const float*)d_in[12];
    const float* W21 = (const float*)d_in[13], *B21 = (const float*)d_in[14];
    const float* W22 = (const float*)d_in[15], *B22 = (const float*)d_in[16];
    const float* W30 = (const float*)d_in[17], *B30 = (const float*)d_in[18];
    const float* W31 = (const float*)d_in[19], *B31 = (const float*)d_in[20];
    const float* W32 = (const float*)d_in[21], *B32 = (const float*)d_in[22];
    float* out = (float*)d_out;

    char* ws = (char*)d_ws;
    size_t off = 0;
    auto alloc = [&](size_t bytes) {
        size_t o = off;
        off += (bytes + 255) & ~(size_t)255;
        return o;
    };
    _Float16* w10f = (_Float16*)(ws + alloc(128 * 64 * 2));
    _Float16* w11f = (_Float16*)(ws + alloc(64 * 64 * 2));
    _Float16* w12f = (_Float16*)(ws + alloc(64 * 128 * 2));
    _Float16* w20f = (_Float16*)(ws + alloc(160 * 128 * 2));
    _Float16* w21f = (_Float16*)(ws + alloc(128 * 128 * 2));
    _Float16* w22f = (_Float16*)(ws + alloc(128 * 256 * 2));
    _Float16* w30f = (_Float16*)(ws + alloc(288 * 256 * 2));
    _Float16* w31f = (_Float16*)(ws + alloc(256 * 512 * 2));
    _Float16* w32f = (_Float16*)(ws + alloc(512 * 1024 * 2));
    int*      nbrp  = (int*)(ws + alloc((size_t)16 * 512 * 64 * 4));
    float*    ndst  = (float*)(ws + alloc((size_t)16 * 512 * 64 * 4));
    _Float16* x1p   = (_Float16*)(ws + alloc((size_t)16 * 512 * 128 * 2));
    _Float16* x2p   = (_Float16*)(ws + alloc((size_t)16 * 512 * 256 * 2));

    zero_out<<<(16 * 1024 + 255) / 256, 256, 0, stream>>>(out, 16 * 1024);

    auto prep = [&](const float* W, _Float16* F, int K, int N, int Kpad) {
        int total = Kpad * N;
        prep_weight<<<(total + 255) / 256, 256, 0, stream>>>(W, F, K, N, Kpad);
    };
    prep(W10, w10f, 101, 64, 128);
    prep(W11, w11f, 64, 64, 64);
    prep(W12, w12f, 64, 128, 64);
    prep(W20, w20f, 131, 128, 160);
    prep(W21, w21f, 128, 128, 128);
    prep(W22, w22f, 128, 256, 128);
    prep(W30, w30f, 259, 256, 288);
    prep(W31, w31f, 256, 512, 256);
    prep(W32, w32f, 512, 1024, 512);

    topk_kernel<<<16 * 512, 256, 0, stream>>>(frac, trans, nbrp, ndst);

    sa_kernel<float, 98, 128, 64, 64, 128><<<16 * 512, 128, 0, stream>>>(
        x, pos, nbrp, ndst, scale, 0.35f, w10f, B10, w11f, B11, w12f, B12, x1p);

    sa_kernel<_Float16, 128, 160, 128, 128, 256><<<16 * 512, 128, 0, stream>>>(
        x1p, pos, nbrp, ndst, scale, 0.5f, w20f, B20, w21f, B21, w22f, B22, x2p);

    mlp3_kernel<<<16 * 32, 128, 0, stream>>>(x2p, pos, w30f, B30, w31f, B31,
                                             w32f, B32, out);
}
